// ProjConvT2IAgg_87746181857683
// MI455X (gfx1250) — matrix operand order repaired and checked by reading, not compiled, not run on hardware
//
#include <hip/hip_runtime.h>
#include <hip/hip_bf16.h>

typedef __attribute__((ext_vector_type(16))) __bf16 v16bf;
typedef __attribute__((ext_vector_type(8)))  __bf16 v8bf;
typedef __attribute__((ext_vector_type(4)))  __bf16 v4bf;
typedef __attribute__((ext_vector_type(8)))  float  v8f;
typedef __attribute__((ext_vector_type(4)))  unsigned int u32x4;
typedef __attribute__((ext_vector_type(8)))  unsigned int u32x8;

#define LATENT   1024
#define GROUPS   16
#define CPG      64
#define KW       3
#define REGIONS  36
#define QDIM     256
#define NW       (LATENT*CPG*KW)   /* 196608 */
#define BI       128
#define BC       32
#define TCAP     50
#define NEGS     0.1f

#define IMGROWS  50   /* padded LDS image slice: row 0 = pad(-1), rows 1..36 = data, 37..49 = pad */

__device__ __forceinline__ float leaky(float x) { return x > 0.f ? x : NEGS * x; }

__device__ __forceinline__ v16bf cat8(v8bf lo, v8bf hi) {
    return __builtin_shufflevector(lo, hi, 0,1,2,3,4,5,6,7,8,9,10,11,12,13,14,15);
}

// ---------------------------------------------------------------------------
// TDM: DMA a 36-row x 64-channel bf16 tile (row stride 1024 elems) into LDS.
// D# per CDNA5 ISA ch.8: group0 = {count, lds_addr, global_addr, type=2},
// group1 = {mask=0, data_size=2B, tensor_dim0=64, tensor_dim1=36,
//           tile_dim0=64, tile_dim1=36, tensor_dim0_stride=1024}.
// Issued by one wave; completion tracked with TENSORcnt.
// ---------------------------------------------------------------------------
__device__ __forceinline__ void tdm_load_slice(const __bf16* gsrc, __bf16* lds_dst)
{
    const unsigned long long ga = (unsigned long long)(size_t)gsrc;
    const unsigned int la = (unsigned int)(size_t)lds_dst;   // addr[31:0] = LDS offset
    u32x4 g0;
    g0[0] = 1u;                                               // count=1, user mode
    g0[1] = la;                                               // lds_addr [63:32]
    g0[2] = (unsigned int)ga;                                 // global_addr low
    g0[3] = (unsigned int)((ga >> 32) & 0x1FFFFFFu)           // global_addr [120:96]
          | (2u << 30);                                       // type=2 ("image")
    u32x8 g1;
    g1[0] = 0x00010000u;          // workgroup_mask=0 (not in cluster), data_size=1 -> 2B
    g1[1] = (unsigned int)CPG << 16;      // tensor_dim0[15:0] at bits 63:48
    g1[2] = (unsigned int)REGIONS << 16;  // tensor_dim0 hi=0; tensor_dim1[15:0] at 95:80
    g1[3] = (unsigned int)CPG << 16;      // tensor_dim1 hi=0; tile_dim0 at 127:112
    g1[4] = (unsigned int)REGIONS;        // tile_dim1 at 143:128; tile_dim2=0
    g1[5] = (unsigned int)LATENT;         // tensor_dim0_stride low 32 (bits 191:160)
    g1[6] = 0u;                           // stride hi=0; tensor_dim1_stride lo=0
    g1[7] = 0u;
    asm volatile("tensor_load_to_lds %0, %1" :: "s"(g0), "s"(g1) : "memory");
}

// ---------------------------------------------------------------------------
// 1) cap_repr (masked mean), cap_act, l2norm(cap_repr), q = cap_repr @ red_w
// ---------------------------------------------------------------------------
__global__ void cap_kernel(const float* __restrict__ cap_embed,
                           const int*   __restrict__ lens,
                           const float* __restrict__ red_w,
                           const float* __restrict__ red_b,
                           float* __restrict__ cap_repr,
                           float* __restrict__ cap_act,
                           float* __restrict__ capv,
                           float* __restrict__ q)
{
    __shared__ float repr[LATENT];
    __shared__ float red[256];
    const int n = blockIdx.x, tid = threadIdx.x;
    const int len = lens[n];
    const float inv = 1.f / (float)len;
    for (int c = tid; c < LATENT; c += 256) {
        float s = 0.f;
        const float* p = cap_embed + (size_t)n * TCAP * LATENT + c;
        for (int t = 0; t < len; ++t) s += p[t * LATENT];
        s *= inv;
        repr[c] = s;
        cap_repr[n * LATENT + c] = s;
        cap_act [n * LATENT + c] = leaky(s);
    }
    __syncthreads();
    float ss = 0.f;
    for (int c = tid; c < LATENT; c += 256) { float v = repr[c]; ss += v * v; }
    red[tid] = ss; __syncthreads();
    for (int s = 128; s > 0; s >>= 1) {
        if (tid < s) red[tid] += red[tid + s];
        __syncthreads();
    }
    const float rnorm = rsqrtf(red[0]);
    for (int c = tid; c < LATENT; c += 256) capv[n * LATENT + c] = repr[c] * rnorm;
    // q: one output per thread (QDIM == 256 == blockDim)
    float acc = red_b[tid];
    for (int c = 0; c < LATENT; ++c) acc += repr[c] * red_w[c * QDIM + tid];
    q[n * QDIM + tid] = acc;
}

// ---------------------------------------------------------------------------
// 2) wraw = q @ proj_w + proj_b   (32 x 196608 x 256) via bf16 WMMA.
//    One wave owns a 16-column tile and BOTH 16-row M-tiles, so each B
//    fragment (proj_w, the 201 MB stream) is loaded exactly once.
// ---------------------------------------------------------------------------
__global__ void __launch_bounds__(128)
wraw_kernel(const float* __restrict__ q,
            const float* __restrict__ proj_w,
            const float* __restrict__ proj_b,
            float* __restrict__ wraw)
{
    const int wave = threadIdx.x >> 5;
    const int lane = threadIdx.x & 31;
    const int hi   = lane >> 4;
    const int ln   = lane & 15;
    const int nt   = blockIdx.x * 4 + wave;
    const int ncol = nt * 16 + ln;

    v8f c0 = {}, c1 = {};
    for (int ks = 0; ks < 8; ++ks) {
        // B fragment: 16 halves = rows K = ks*32 + hi*16 + [0..16), column ncol
        v16bf b;
        for (int h = 0; h < 16; ++h)
            b[h] = (__bf16)proj_w[(size_t)(ks * 32 + hi * 16 + h) * NW + ncol];
        // A fragments for both M-tiles: per-lane contiguous 8-float chunks
        const int kb0 = ks * 32 + hi * 16;
        const int kb1 = kb0 + 8;
        v16bf a0, a1;
        for (int h = 0; h < 8; ++h) {
            a0[h]     = (__bf16)q[ln * QDIM + kb0 + h];
            a0[h + 8] = (__bf16)q[ln * QDIM + kb1 + h];
            a1[h]     = (__bf16)q[(16 + ln) * QDIM + kb0 + h];
            a1[h + 8] = (__bf16)q[(16 + ln) * QDIM + kb1 + h];
        }
        c0 = __builtin_amdgcn_wmma_f32_16x16x32_bf16(false, a0, false, b,
                                                     (short)0, c0, false, false);
        c1 = __builtin_amdgcn_wmma_f32_16x16x32_bf16(false, a1, false, b,
                                                     (short)0, c1, false, false);
    }
    const float bias = proj_b[ncol];
    for (int v = 0; v < 8; ++v) {
        const int r0 = v + (hi ? 8 : 0);         // C layout: row = v (+8 for hi lanes)
        wraw[(size_t)r0 * NW + ncol]        = c0[v] + bias;
        wraw[(size_t)(16 + r0) * NW + ncol] = c1[v] + bias;
    }
}

// ---------------------------------------------------------------------------
// 3) softmax over K=3 + relayout to bf16 [n][g][o][k*64+i]
// ---------------------------------------------------------------------------
__global__ void softmax_kernel(const float* __restrict__ wraw,
                               __bf16* __restrict__ Wbuf)
{
    const int idx   = blockIdx.x * 256 + threadIdx.x;   // 32*1024*64 = 2097152
    const int i     = idx & 63;
    const int outch = (idx >> 6) & 1023;
    const int n     = idx >> 16;
    const size_t base = (size_t)n * NW + (size_t)outch * (CPG * KW) + i * KW;
    const float w0 = wraw[base], w1 = wraw[base + 1], w2 = wraw[base + 2];
    const float m  = fmaxf(w0, fmaxf(w1, w2));
    const float e0 = __expf(w0 - m), e1 = __expf(w1 - m), e2 = __expf(w2 - m);
    const float rs = 1.f / (e0 + e1 + e2);
    const int g = outch >> 6, o = outch & 63;
    const size_t wb = (((size_t)n * GROUPS + g) * CPG + o) * (KW * CPG);
    Wbuf[wb + 0 * 64 + i] = (__bf16)(e0 * rs);
    Wbuf[wb + 1 * 64 + i] = (__bf16)(e1 * rs);
    Wbuf[wb + 2 * 64 + i] = (__bf16)(e2 * rs);
}

// ---------------------------------------------------------------------------
// 4) img f32 -> bf16 (vectorized: float4 in, 4x bf16 out)
// ---------------------------------------------------------------------------
__global__ void imgcvt_kernel(const float4* __restrict__ img,
                              v4bf* __restrict__ imgbf, int nvec)
{
    const int idx = blockIdx.x * 256 + threadIdx.x;
    if (idx < nvec) {
        const float4 f = img[idx];
        v4bf o;
        o[0] = (__bf16)f.x; o[1] = (__bf16)f.y;
        o[2] = (__bf16)f.z; o[3] = (__bf16)f.w;
        imgbf[idx] = o;
    }
}

// ---------------------------------------------------------------------------
// 5) dynamic grouped conv via bf16 WMMA + fused leaky + mean over regions.
//    TDM double-buffers the (b,g) image slice into LDS (wave 0 issues the
//    DMA for slice b+1 while all waves run WMMAs on slice b); zero-padded
//    rows make B fragments branch-free 16-byte LDS vector loads.
//    grid: (n=32, g=16, 8 image-chunks), 128 threads = 4 waves (M-tiles)
// ---------------------------------------------------------------------------
__global__ void __launch_bounds__(128)
conv_kernel(const __bf16* __restrict__ Wbuf,
            const __bf16* __restrict__ imgbf,
            float* __restrict__ Mout)
{
    __shared__ __bf16 Wl[CPG * 192];            // 24576 B: A matrix 64x192
    __shared__ __bf16 imgl[2][IMGROWS * CPG];   // 12800 B: double-buffered slice
    __shared__ float  wc[4][16][48];            // 12288 B: per-wave C tiles

    const int n = blockIdx.x, g = blockIdx.y, bcnk = blockIdx.z;
    const int tid  = threadIdx.x;
    const int wave = tid >> 5, lane = tid & 31, hi = lane >> 4, ln = lane & 15;

    // stage W[n][g] (64x192 bf16) as 16B vectors
    {
        const uint4* src = (const uint4*)(Wbuf + ((size_t)n * GROUPS + g) * CPG * 192);
        uint4* dst = (uint4*)Wl;
        for (int j = tid; j < CPG * 192 / 8; j += 128) dst[j] = src[j];
    }
    // zero pad rows (row 0 and rows 37..49) of both buffers once;
    // disjoint from the rows the TDM writes (1..36), so no race with the DMA
    for (int buf = 0; buf < 2; ++buf)
        for (int j = tid; j < IMGROWS * CPG; j += 128) {
            const int row = j >> 6;
            if (row == 0 || row > REGIONS) imgl[buf][j] = (__bf16)0.f;
        }
    // kick off the first slice DMA (wave 0 only; TENSORcnt-tracked)
    const int b0 = bcnk * 16;
    if (tid < 32) {
        tdm_load_slice(imgbf + (size_t)b0 * REGIONS * LATENT + g * CPG, &imgl[0][CPG]);
        __builtin_amdgcn_s_wait_tensorcnt(0);
    }
    __syncthreads();

    int cur = 0;
    for (int bb = 0; bb < 16; ++bb) {
        const int b = b0 + bb;
        // prefetch next slice into the other buffer via TDM (overlaps compute)
        if (tid < 32 && bb + 1 < 16)
            tdm_load_slice(imgbf + (size_t)(b + 1) * REGIONS * LATENT + g * CPG,
                           &imgl[cur ^ 1][CPG]);
        const __bf16* im = imgl[cur];

        const int mt = wave;                   // wave owns output rows [mt*16, mt*16+16)
        const int orow = mt * 16 + ln;
        v8f c0 = {}, c1 = {}, c2 = {};
        for (int ks = 0; ks < 6; ++ks) {       // K = 192 = 6 x 32
            // A fragment: two contiguous 8-half chunks per lane
            const int kb0 = ks * 32 + hi * 16;
            const v8bf alo = *(const v8bf*)&Wl[orow * 192 + kb0];
            const v8bf ahi = *(const v8bf*)&Wl[orow * 192 + kb0 + 8];
            const v16bf a = cat8(alo, ahi);
            // B fragments: 16 contiguous halves per lane; rows r=ks*32+hi*16+[0..16)
            // stay inside one k-block of 64, so addr = im[(l + k)*64 + iib]
            const int rbase = ks * 32 + hi * 16;
            const int k     = rbase >> 6;
            const int iib   = rbase & 63;
            const v16bf bf0 = *(const v16bf*)&im[(ln +  0 + k) * CPG + iib];
            const v16bf bf1 = *(const v16bf*)&im[(ln + 16 + k) * CPG + iib];
            const v16bf bf2 = *(const v16bf*)&im[(ln + 32 + k) * CPG + iib];
            c0 = __builtin_amdgcn_wmma_f32_16x16x32_bf16(false, a, false, bf0,
                                                         (short)0, c0, false, false);
            c1 = __builtin_amdgcn_wmma_f32_16x16x32_bf16(false, a, false, bf1,
                                                         (short)0, c1, false, false);
            c2 = __builtin_amdgcn_wmma_f32_16x16x32_bf16(false, a, false, bf2,
                                                         (short)0, c2, false, false);
        }
        for (int v = 0; v < 8; ++v) {
            const int Mr = v + (hi ? 8 : 0);
            wc[mt][Mr][ 0 + ln] = c0[v];
            wc[mt][Mr][16 + ln] = c1[v];
            wc[mt][Mr][32 + ln] = c2[v];
        }
        __syncthreads();
        if (tid < 64) {
            const int o = tid;
            float s = 0.f;
            for (int l = 0; l < REGIONS; ++l) s += leaky(wc[o >> 4][o & 15][l]);
            Mout[((size_t)n * BI + b) * LATENT + g * CPG + o] = s * (1.f / REGIONS);
        }
        // make sure the prefetched slice has landed before anyone reads it
        if (tid < 32) __builtin_amdgcn_s_wait_tensorcnt(0);
        __syncthreads();
        cur ^= 1;
    }
}

// ---------------------------------------------------------------------------
// 6) sims[b,n] = (v . capv_n) / ||v||,  v = M[n,b,:] * cap_act[n,:]
// ---------------------------------------------------------------------------
__global__ void final_kernel(const float* __restrict__ Mout,
                             const float* __restrict__ cap_act,
                             const float* __restrict__ capv,
                             float* __restrict__ out)
{
    __shared__ float r1[256], r2[256];
    const int b = blockIdx.x, n = blockIdx.y, tid = threadIdx.x;
    const float* mv = Mout    + ((size_t)n * BI + b) * LATENT;
    const float* ca = cap_act + n * LATENT;
    const float* cv = capv    + n * LATENT;
    float n2 = 0.f, dt = 0.f;
    for (int c = tid; c < LATENT; c += 256) {
        const float v = mv[c] * ca[c];
        n2 += v * v;
        dt += v * cv[c];
    }
    r1[tid] = n2; r2[tid] = dt; __syncthreads();
    for (int s = 128; s > 0; s >>= 1) {
        if (tid < s) { r1[tid] += r1[tid + s]; r2[tid] += r2[tid + s]; }
        __syncthreads();
    }
    if (tid == 0) out[b * BC + n] = r2[0] * rsqrtf(r1[0]);
}

// ---------------------------------------------------------------------------
extern "C" void kernel_launch(void* const* d_in, const int* in_sizes, int n_in,
                              void* d_out, int out_size, void* d_ws, size_t ws_size,
                              hipStream_t stream)
{
    const float* img_embed = (const float*)d_in[0];
    const float* cap_embed = (const float*)d_in[1];
    const int*   lens      = (const int*)  d_in[2];
    const float* red_w     = (const float*)d_in[3];
    const float* red_b     = (const float*)d_in[4];
    const float* proj_w    = (const float*)d_in[5];
    const float* proj_b    = (const float*)d_in[6];
    float* out = (float*)d_out;

    char* ws = (char*)d_ws;
    float*  cap_repr = (float*)(ws + 0x00000);                 // 128 KB
    float*  cap_act  = (float*)(ws + 0x20000);                 // 128 KB
    float*  capv     = (float*)(ws + 0x40000);                 // 128 KB
    float*  qbuf     = (float*)(ws + 0x60000);                 //  32 KB
    float*  wraw     = (float*)(ws + 0x70000);                 // 25.2 MB (reused as Mout)
    __bf16* Wbuf     = (__bf16*)(ws + 0x70000 + 25165824);     // 12.6 MB
    __bf16* imgbf    = (__bf16*)(ws + 0x70000 + 25165824 + 12582912); // 9.4 MB
    float*  Mout     = wraw;   // safe: wraw fully consumed by softmax_kernel first

    cap_kernel    <<<32, 256, 0, stream>>>(cap_embed, lens, red_w, red_b,
                                           cap_repr, cap_act, capv, qbuf);
    wraw_kernel   <<<3072, 128, 0, stream>>>(qbuf, proj_w, proj_b, wraw);
    softmax_kernel<<<8192, 256, 0, stream>>>(wraw, Wbuf);
    imgcvt_kernel <<<4608, 256, 0, stream>>>((const float4*)img_embed,
                                             (v4bf*)imgbf, BI * REGIONS * LATENT / 4);
    conv_kernel   <<<dim3(32, 16, 8), 128, 0, stream>>>(Wbuf, imgbf, Mout);
    final_kernel  <<<dim3(128, 32), 256, 0, stream>>>(Mout, cap_act, capv, out);
}
